// SelfAttention_65481071401353
// MI455X (gfx1250) — compile-verified, matrix operand order checked by reading, not run on hardware
//
#include <hip/hip_runtime.h>

// ---------------------------------------------------------------------------
// Self-attention, MI455X (gfx1250, wave32). Compute-bound (~43 GFLOP vs ~20MB
// HBM traffic), so all GEMMs run on v_wmma_f32_16x16x32_bf16 (f32 accumulate).
// Flash-attention structure with double-buffered async global->LDS staging of
// K/V tiles (ASYNCcnt path); GEMM weight slices staged the same way.
// ---------------------------------------------------------------------------

typedef __bf16 bf16;
typedef __attribute__((ext_vector_type(16))) __bf16 v16bf;
typedef __attribute__((ext_vector_type(8)))  float   v8f;
typedef __attribute__((ext_vector_type(4)))  float   v4f;

#define T_SEQ 4096
#define EMB   512
#define NH    8
#define HD    64

union FragU { v16bf v; v4f q[2]; };

// Load a 16x32 bf16 WMMA fragment (A-layout; also serves as B when the B
// matrix's columns are contiguous in memory, i.e. source stored [N][K]).
// ISA layout: row = lane%16; K-base = (lane>=16)?8:0; element j -> K =
// base + j + 8*(j>=8)  => two contiguous 8-elem (16B) chunks at +0 and +16.
__device__ __forceinline__ v16bf load_frag(const bf16* base, int stride,
                                           int row0, int k0) {
  const int lane = threadIdx.x & 31;
  const int r    = lane & 15;
  const int kb   = (lane >> 4) << 3;   // 0 or 8
  const bf16* p = base + (size_t)(row0 + r) * stride + k0 + kb;
  FragU u;
  u.q[0] = *(const v4f*)(const void*)(p);
  u.q[1] = *(const v4f*)(const void*)(p + 16);
  return u.v;
}

__device__ __forceinline__ v8f wmma_bf16(v16bf a, v16bf b, v8f c) {
  return __builtin_amdgcn_wmma_f32_16x16x32_bf16(
      /*neg_a=*/false, a, /*neg_b=*/false, b,
      /*c_mod=*/(short)0, c, /*reuse_a=*/false, /*reuse_b=*/false);
}

// One 16B async global->LDS copy (GV mode, per-lane LDS byte address in VGPR;
// generic-pointer low 32 bits are the LDS offset per the aperture rules).
__device__ __forceinline__ void async_copy_b128(const bf16* g, void* l) {
  asm volatile("global_load_async_to_lds_b128 %0, %1, off"
               :: "v"((unsigned)(uintptr_t)l), "v"(g) : "memory");
}

// ---------------------------------------------------------------------------
// Kernel 1: f32 -> bf16 conversion.
// ---------------------------------------------------------------------------
__global__ void cvt_f32_bf16(const float* __restrict__ s, bf16* __restrict__ d,
                             int n) {
  int i  = blockIdx.x * blockDim.x + threadIdx.x;
  int st = gridDim.x * blockDim.x;
  for (; i < n; i += st) d[i] = (bf16)s[i];
}

// ---------------------------------------------------------------------------
// Kernel 2: y = A[4096x512] @ W^T (W stored [out=512][in=512] row-major, so
// B columns == W rows == contiguous). Block = 8 waves sharing one 64-wide
// N-group; the 64x512 bf16 W slice (64KB) is staged once into LDS with
// GLOBAL_LOAD_ASYNC_TO_LDS_B128 (ASYNCcnt). Each wave computes 32(M) x 64(N):
// per K-step 2 A-frags (global b128) x 4 B-frags (LDS b128) -> 8 WMMAs.
// Store layout by mode:
//   0: Q  -> [H][T][64] bf16, scaled by 1/sqrt(HD)
//   1: K  -> [H][T][64] bf16
//   2: V  -> [H][64][T] bf16 (transposed so PV B-fragments are contiguous)
//   3: O  -> [T][512] f32 (final output)
// ---------------------------------------------------------------------------
__global__ void __launch_bounds__(256)
gemm_proj(const bf16* __restrict__ A, const bf16* __restrict__ W,
          bf16* __restrict__ outb, float* __restrict__ outf, int mode) {
  __shared__ bf16 Wl[64][520];                    // +8 bf16 pad per row

  const int wid  = threadIdx.x >> 5;
  const int lane = threadIdx.x & 31;
  const int n0 = (blockIdx.x & 7) * 64;           // 8 N-groups
  const int m0 = (blockIdx.x >> 3) * 256 + wid * 32;  // 16 M-blocks x 8 waves

  // ---- async-stage W[n0..n0+64) x [0..512) into LDS (16 b128 per lane) ----
  for (int e = threadIdx.x; e < 4096; e += 256) {
    const int row = e >> 6;                       // 64 chunks of 8 per row
    const int col = (e & 63) << 3;
    async_copy_b128(W + (size_t)(n0 + row) * EMB + col, &Wl[row][col]);
  }
  asm volatile("s_wait_asynccnt 0x0" ::: "memory");
  __syncthreads();

  v8f acc[2][4] = {};
  for (int k0 = 0; k0 < EMB; k0 += 32) {
    v16bf a0 = load_frag(A, EMB, m0,      k0);
    v16bf a1 = load_frag(A, EMB, m0 + 16, k0);
#pragma unroll
    for (int t = 0; t < 4; ++t) {
      v16bf b = load_frag(&Wl[0][0], 520, t * 16, k0);
      acc[0][t] = wmma_bf16(a0, b, acc[0][t]);
      acc[1][t] = wmma_bf16(a1, b, acc[1][t]);
    }
  }

  const int N  = lane & 15;
  const int Mb = (lane >> 4) * 8;
#pragma unroll
  for (int im = 0; im < 2; ++im) {
#pragma unroll
    for (int t = 0; t < 4; ++t) {
#pragma unroll
      for (int r = 0; r < 8; ++r) {
        const int m = m0 + im * 16 + Mb + r;
        const int f = n0 + t * 16 + N;
        const float y = acc[im][t][r];
        if (mode == 0) {
          int h = f >> 6, d = f & 63;
          outb[((size_t)h * T_SEQ + m) * HD + d] = (bf16)(y * 0.125f);
        } else if (mode == 1) {
          int h = f >> 6, d = f & 63;
          outb[((size_t)h * T_SEQ + m) * HD + d] = (bf16)y;
        } else if (mode == 2) {
          int h = f >> 6, d = f & 63;
          outb[((size_t)h * HD + d) * T_SEQ + m] = (bf16)y;   // V^T
        } else {
          outf[(size_t)m * EMB + f] = y;
        }
      }
    }
  }
}

// ---------------------------------------------------------------------------
// Kernel 3: flash attention. Block = 4 waves, all on the SAME head and key
// range (different 16-query tiles), so K/V tiles are staged cooperatively
// into LDS with double-buffered async copies (4 b128 per thread per stage ->
// s_wait_asynccnt 4 retires exactly the current tile while the prefetch for
// the next tile stays in flight behind the WMMAs).
// Per 32-key step: 4 score WMMAs (QK^T from LDS), online softmax with
// shfl_xor row reductions, P re-swizzled C-layout -> A-layout via LDS (bf16),
// 4 PV WMMAs (V^T from LDS). Q already carries the 1/sqrt(HD) scale.
// ---------------------------------------------------------------------------
__global__ void __launch_bounds__(128)
attn_kernel(const bf16* __restrict__ Q, const bf16* __restrict__ K,
            const bf16* __restrict__ Vt, bf16* __restrict__ out) {
  __shared__ bf16 Kl[2][32][72];                 // 2 x 4.5 KB, row = 144B
  __shared__ bf16 Vl[2][64][40];                 // 2 x 5  KB, row =  80B
  __shared__ bf16 Pl[4][16][32];                 // 1 KB per wave

  const int wid    = threadIdx.x >> 5;
  const int lane   = threadIdx.x & 31;
  const int waveId = blockIdx.x * 4 + wid;       // 2048 waves
  const int h  = waveId >> 8;                    // 0..7 (uniform per block)
  const int q0 = (waveId & 255) * 16;            // query tile base

  const bf16* Qh = Q  + (size_t)h * T_SEQ * HD;
  const bf16* Kh = K  + (size_t)h * T_SEQ * HD;
  const bf16* Vh = Vt + (size_t)h * HD * T_SEQ;  // [64][T]
  bf16* Pw = &Pl[wid][0][0];

  const int Mb = (lane >> 4) * 8;

  // Cooperative async stage of one 32-key tile: K 32x64, V^T 64x32 (bf16).
  auto stage_kv = [&](int buf, int s0) {
    for (int c = threadIdx.x; c < 256; c += 128) {       // K: 256 b128 chunks
      const int row = c >> 3, col = (c & 7) << 3;
      async_copy_b128(Kh + (size_t)(s0 + row) * HD + col, &Kl[buf][row][col]);
    }
    for (int c = threadIdx.x; c < 256; c += 128) {       // V: 256 b128 chunks
      const int row = c >> 2, col = (c & 3) << 3;
      async_copy_b128(Vh + (size_t)row * T_SEQ + s0 + col, &Vl[buf][row][col]);
    }
  };

  v8f o[4] = {};
  float rm[8], rs[8];
#pragma unroll
  for (int r = 0; r < 8; ++r) { rm[r] = -1e30f; rs[r] = 0.f; }

  // Q fragments are reused across the whole key loop.
  const v16bf aq0 = load_frag(Qh, HD, q0, 0);
  const v16bf aq1 = load_frag(Qh, HD, q0, 32);

  stage_kv(0, 0);
  int buf = 0;

  for (int s0 = 0; s0 < T_SEQ; s0 += 32) {
    // Prefetch next tile into the other buffer, then retire current tile's
    // 4 per-thread async copies (in-order completion).
    if (s0 + 32 < T_SEQ) {
      stage_kv(buf ^ 1, s0 + 32);
      asm volatile("s_wait_asynccnt 0x4" ::: "memory");
    } else {
      asm volatile("s_wait_asynccnt 0x0" ::: "memory");
    }
    __syncthreads();                             // staged data visible to all

    const bf16* Kt = &Kl[buf][0][0];
    const bf16* Vb = &Vl[buf][0][0];

    // ---- scores: S[16 x 32] = Q(16x64) @ K^T, two 16-col halves ----
    v8f sc0 = {}, sc1 = {};
    sc0 = wmma_bf16(aq0, load_frag(Kt, 72, 0,  0),  sc0);
    sc0 = wmma_bf16(aq1, load_frag(Kt, 72, 0,  32), sc0);
    sc1 = wmma_bf16(aq0, load_frag(Kt, 72, 16, 0),  sc1);
    sc1 = wmma_bf16(aq1, load_frag(Kt, 72, 16, 32), sc1);

    // ---- online softmax (row = Mb + r, spread over 16 lanes) ----
    float sa[8];
#pragma unroll
    for (int r = 0; r < 8; ++r) {
      float t = fmaxf(sc0[r], sc1[r]);
      t = fmaxf(t, __shfl_xor(t, 1, 32));
      t = fmaxf(t, __shfl_xor(t, 2, 32));
      t = fmaxf(t, __shfl_xor(t, 4, 32));
      t = fmaxf(t, __shfl_xor(t, 8, 32));
      const float nm = fmaxf(rm[r], t);
      const float sc = __expf(rm[r] - nm);
      rm[r] = nm;
      rs[r] *= sc;
      sa[r]  = sc;

      const float p0 = __expf(sc0[r] - nm);
      const float p1 = __expf(sc1[r] - nm);
      const int M = Mb + r;
      Pw[M * 32 + (lane & 15)]      = (bf16)p0;
      Pw[M * 32 + 16 + (lane & 15)] = (bf16)p1;

      float ps = p0 + p1;
      ps += __shfl_xor(ps, 1, 32);
      ps += __shfl_xor(ps, 2, 32);
      ps += __shfl_xor(ps, 4, 32);
      ps += __shfl_xor(ps, 8, 32);
      rs[r] += ps;
    }

    // rescale running output by exp(m_old - m_new)
#pragma unroll
    for (int t = 0; t < 4; ++t)
#pragma unroll
      for (int r = 0; r < 8; ++r) o[t][r] = o[t][r] * sa[r];

    // P writes are wave-private; wait for ds stores, then re-read P in the
    // 16-bit A-fragment layout.
    __asm__ volatile("s_wait_dscnt 0" ::: "memory");
    const v16bf pf = load_frag(Pw, 32, 0, 0);

    // ---- O += P(16x32) @ V(32x64) from the staged V^T tile ----
#pragma unroll
    for (int t = 0; t < 4; ++t) {
      v16bf bv = load_frag(Vb, 40, t * 16, 0);
      o[t] = wmma_bf16(pf, bv, o[t]);
    }

    __syncthreads();        // everyone done reading buf before it is restaged
    buf ^= 1;
  }

  // ---- normalize and store interleaved [T][H*64] bf16 ----
#pragma unroll
  for (int r = 0; r < 8; ++r) rs[r] = 1.0f / rs[r];
#pragma unroll
  for (int t = 0; t < 4; ++t)
#pragma unroll
    for (int r = 0; r < 8; ++r) {
      const int tq = q0 + Mb + r;
      const int d  = t * 16 + (lane & 15);
      out[(size_t)tq * EMB + h * HD + d] = (bf16)(o[t][r] * rs[r]);
    }
}

// ---------------------------------------------------------------------------
// Launcher. Workspace layout (22 MiB):
//   0        x_bf16   [4096][512]
//   4  MiB   wq,wk,wv,wo bf16 (512KiB each)
//   6  MiB   Q  [8][4096][64] bf16
//   10 MiB   K  [8][4096][64] bf16
//   14 MiB   Vt [8][64][4096] bf16
//   18 MiB   attn_out [4096][512] bf16
// ---------------------------------------------------------------------------
extern "C" void kernel_launch(void* const* d_in, const int* in_sizes, int n_in,
                              void* d_out, int out_size, void* d_ws,
                              size_t ws_size, hipStream_t stream) {
  (void)in_sizes; (void)n_in; (void)out_size; (void)ws_size;

  const float* x  = (const float*)d_in[0];
  const float* wq = (const float*)d_in[1];
  const float* wk = (const float*)d_in[2];
  const float* wv = (const float*)d_in[3];
  const float* wo = (const float*)d_in[4];

  char* ws = (char*)d_ws;
  bf16* xb  = (bf16*)(ws);
  bf16* wqb = (bf16*)(ws + (4u << 20));
  bf16* wkb = (bf16*)(ws + (4u << 20) + 512u * 1024u);
  bf16* wvb = (bf16*)(ws + (5u << 20));
  bf16* wob = (bf16*)(ws + (5u << 20) + 512u * 1024u);
  bf16* Qb  = (bf16*)(ws + (6u << 20));
  bf16* Kb  = (bf16*)(ws + (10u << 20));
  bf16* Vtb = (bf16*)(ws + (14u << 20));
  bf16* AOb = (bf16*)(ws + (18u << 20));

  const int nX = T_SEQ * EMB, nW = EMB * EMB;
  cvt_f32_bf16<<<256, 256, 0, stream>>>(x,  xb,  nX);
  cvt_f32_bf16<<<64,  256, 0, stream>>>(wq, wqb, nW);
  cvt_f32_bf16<<<64,  256, 0, stream>>>(wk, wkb, nW);
  cvt_f32_bf16<<<64,  256, 0, stream>>>(wv, wvb, nW);
  cvt_f32_bf16<<<64,  256, 0, stream>>>(wo, wob, nW);

  gemm_proj<<<128, 256, 0, stream>>>(xb, wqb, Qb,  nullptr, 0);  // Q (scaled)
  gemm_proj<<<128, 256, 0, stream>>>(xb, wkb, Kb,  nullptr, 1);  // K
  gemm_proj<<<128, 256, 0, stream>>>(xb, wvb, Vtb, nullptr, 2);  // V^T

  attn_kernel<<<512, 128, 0, stream>>>(Qb, Kb, Vtb, AOb);

  gemm_proj<<<128, 256, 0, stream>>>(AOb, wob, nullptr, (float*)d_out, 3);
}
